// MultiscaleGlobalWaveGNN_40037685133395
// MI455X (gfx1250) — matrix operand order repaired and checked by reading, not compiled, not run on hardware
//
#include <hip/hip_runtime.h>
#include <hip/hip_bf16.h>
#include <math.h>

// ---------------------------------------------------------------------------
// MultiscaleGlobalWaveGNN forward for MI455X (gfx1250, wave32, WMMA)
// 64x64 output tile per 128-thread block (4 waves), f16 WMMA, f32 accumulate.
// Staging: unconditional float4 global loads from clamped addresses (no
// divergent per-element guards), f32->f16 pack, ds_store_b64 into LDS.
// ---------------------------------------------------------------------------

typedef __attribute__((ext_vector_type(16))) _Float16 v16h;
typedef __attribute__((ext_vector_type(8)))  float    v8f;

#define TILE_M 64
#define TILE_N 64
#define TILE_K 32
#define LDSD   40   // padded LDS row stride in halfs (80B, 16B-aligned)

union HFrag { v16h v; uint4 q[2]; };
union H4 { _Float16 h[4]; uint2 u; };

__device__ __forceinline__ float actf(float v, int act) {
  if (act == 1) return v > 0.f ? v : 0.f;
  if (act == 2) return 1.f / (1.f + __expf(-v));
  return v;
}

// A fragment (16x32 f16): lane 0-15 row M=lane, K {0..7,16..23}; lane 16-31 K {8..15,24..31}
// B fragment (32x16 f16): lane n col N=n%16, lanes<16 K {0..7,16..23}, lanes>=16 K {8..15,24..31} (Bs: [n][k])
__device__ __forceinline__ void wmma_tile(const _Float16* As, const _Float16* Bs,
                                          int wave, int lane, v8f acc[4]) {
  const int arow = lane & 15;
  const int aoff = (wave * 16 + arow) * LDSD + ((lane < 16) ? 0 : 8);
  HFrag a;
  a.q[0] = *(const uint4*)(As + aoff);
  a.q[1] = *(const uint4*)(As + aoff + 16);
#pragma unroll
  for (int ns = 0; ns < 4; ++ns) {
    const int bcol = ns * 16 + (lane & 15);
    const int boff = bcol * LDSD + ((lane < 16) ? 0 : 8);
    HFrag b;
    b.q[0] = *(const uint4*)(Bs + boff);
    b.q[1] = *(const uint4*)(Bs + boff + 16);
    acc[ns] = __builtin_amdgcn_wmma_f32_16x16x32_f16(false, a.v, false, b.v,
                                                     (short)0, acc[ns], false, false);
  }
}

// --- vectorized staging helpers (all loads in-bounds via clamping) ----------

// A tile (64 rows x 32 k): requires lda%4==0 and K%4==0.
__device__ __forceinline__ void stage_A(const float* __restrict__ A, int lda, int M, int K,
                                        int rowBase, int k0, int tid, _Float16* As) {
  float4 v[4]; int rr[4], qq[4]; bool ok[4];
#pragma unroll
  for (int it = 0; it < 4; ++it) {
    int idx = tid + it * 128;
    int r = idx >> 3, q = (idx & 7) * 4;
    int gr = rowBase + r, gk = k0 + q;
    int grc = gr < M ? gr : M - 1;
    int gkc = gk < K ? gk : 0;
    v[it] = *(const float4*)(A + (size_t)grc * lda + gkc);
    ok[it] = (gr < M) && (gk < K);
    rr[it] = r; qq[it] = q;
  }
#pragma unroll
  for (int it = 0; it < 4; ++it) {
    float4 w = v[it];
    if (!ok[it]) { w.x = 0.f; w.y = 0.f; w.z = 0.f; w.w = 0.f; }
    H4 p;
    p.h[0] = (_Float16)w.x; p.h[1] = (_Float16)w.y;
    p.h[2] = (_Float16)w.z; p.h[3] = (_Float16)w.w;
    *(uint2*)(As + rr[it] * LDSD + qq[it]) = p.u;
  }
}

// B row-major (K,N) tile -> Bs[n][k]; requires N%4==0 (chunks along n).
__device__ __forceinline__ void stage_Bn(const float* __restrict__ B, int N, int K,
                                         int colBase, int k0, int tid, _Float16* Bs) {
  float4 v[4]; int kk_[4], qn[4]; bool ok[4];
#pragma unroll
  for (int it = 0; it < 4; ++it) {
    int idx = tid + it * 128;
    int kk = idx >> 4, q = (idx & 15) * 4;
    int gk = k0 + kk, gn = colBase + q;
    int gkc = gk < K ? gk : K - 1;
    int gnc = (gn + 4 <= N) ? gn : 0;
    v[it] = *(const float4*)(B + (size_t)gkc * N + gnc);
    ok[it] = (gk < K) && (gn + 4 <= N);
    kk_[it] = kk; qn[it] = q;
  }
#pragma unroll
  for (int it = 0; it < 4; ++it) {
    float4 w = v[it];
    if (!ok[it]) { w.x = 0.f; w.y = 0.f; w.z = 0.f; w.w = 0.f; }
    Bs[(qn[it] + 0) * LDSD + kk_[it]] = (_Float16)w.x;
    Bs[(qn[it] + 1) * LDSD + kk_[it]] = (_Float16)w.y;
    Bs[(qn[it] + 2) * LDSD + kk_[it]] = (_Float16)w.z;
    Bs[(qn[it] + 3) * LDSD + kk_[it]] = (_Float16)w.w;
  }
}

// B row-major (N,K) tile (i.e. C = A @ B^T) -> Bs[n][k]; requires K%4==0.
__device__ __forceinline__ void stage_Bt(const float* __restrict__ B, int N, int K,
                                         int colBase, int k0, int tid, _Float16* Bs) {
#pragma unroll
  for (int it = 0; it < 4; ++it) {
    int idx = tid + it * 128;
    int n = idx >> 3, q = (idx & 7) * 4;
    int gn = colBase + n, gk = k0 + q;
    int gnc = gn < N ? gn : N - 1;
    int gkc = gk < K ? gk : 0;
    float4 w = *(const float4*)(B + (size_t)gnc * K + gkc);
    if (!((gn < N) && (gk < K))) { w.x = 0.f; w.y = 0.f; w.z = 0.f; w.w = 0.f; }
    H4 p;
    p.h[0] = (_Float16)w.x; p.h[1] = (_Float16)w.y;
    p.h[2] = (_Float16)w.z; p.h[3] = (_Float16)w.w;
    *(uint2*)(Bs + n * LDSD + q) = p.u;
  }
}

// Generic GEMM: C = act(A(M,K) @ B + bias + beta*C).  TRANSB: B is (N,K).
template <int TRANSB>
__global__ void __launch_bounds__(128) gemm_kernel(
    const float* __restrict__ A, int lda,
    const float* __restrict__ B,
    const float* __restrict__ bias,
    float* __restrict__ C, int ldc,
    int M, int N, int K, int act, int beta)
{
  __shared__ __align__(16) _Float16 As[TILE_M * LDSD];
  __shared__ __align__(16) _Float16 Bs[TILE_N * LDSD];
  const int tid = threadIdx.x;
  const int wave = tid >> 5, lane = tid & 31;
  const int rowBase = blockIdx.y * TILE_M;
  const int colBase = blockIdx.x * TILE_N;
  v8f acc[4] = {};

  for (int k0 = 0; k0 < K; k0 += TILE_K) {
    stage_A(A, lda, M, K, rowBase, k0, tid, As);
    if (TRANSB) stage_Bt(B, N, K, colBase, k0, tid, Bs);
    else        stage_Bn(B, N, K, colBase, k0, tid, Bs);
    __syncthreads();
    wmma_tile(As, Bs, wave, lane, acc);
    __syncthreads();
  }

  const int colL = lane & 15;
  const int rsel = (lane >> 4) * 8;
#pragma unroll
  for (int ns = 0; ns < 4; ++ns) {
    int gcol = colBase + ns * 16 + colL;
    if (gcol >= N) continue;
    float bv = bias ? bias[gcol] : 0.f;
#pragma unroll
    for (int r = 0; r < 8; ++r) {
      int grow = rowBase + wave * 16 + rsel + r;
      if (grow < M) {
        float v = acc[ns][r] + bv;
        if (beta) v += C[(size_t)grow * ldc + gcol];
        C[(size_t)grow * ldc + gcol] = actf(v, act);
      }
    }
  }
}

// Edge message layer 1: A row e = [hn[src[e]](128) | hn[tgt[e]](128) | eattr[e](3)]
// Cout(Ei,128) = relu(A @ W(259,128) + bias)
__global__ void __launch_bounds__(128) edge_msg1_kernel(
    const float* __restrict__ hn,
    const int* __restrict__ src, const int* __restrict__ tgt,
    const float* __restrict__ eattr,
    int eoff, int Ei,
    const float* __restrict__ W, const float* __restrict__ bias,
    float* __restrict__ Cout)
{
  __shared__ __align__(16) _Float16 As[TILE_M * LDSD];
  __shared__ __align__(16) _Float16 Bs[TILE_N * LDSD];
  __shared__ int sIdx[TILE_M], tIdx[TILE_M];
  const int tid = threadIdx.x;
  const int wave = tid >> 5, lane = tid & 31;
  const int rowBase = blockIdx.y * TILE_M;
  const int colBase = blockIdx.x * TILE_N;   // N = 128 -> blockIdx.x in {0,1}
  if (tid < TILE_M) {
    int e = rowBase + tid;
    int eg = eoff + ((e < Ei) ? e : 0);      // clamped -> always valid gather
    sIdx[tid] = src[eg];
    tIdx[tid] = tgt[eg];
  }
  __syncthreads();
  v8f acc[4] = {};

  // fast tiles: k = 0..255 (hn[src] then hn[tgt]), float4 gathers
  for (int k0 = 0; k0 < 256; k0 += TILE_K) {
#pragma unroll
    for (int it = 0; it < 4; ++it) {
      int idx = tid + it * 128;
      int r = idx >> 3, q = (idx & 7) * 4;
      int gk = k0 + q;
      const float* base = (gk < 128) ? (hn + (size_t)sIdx[r] * 128 + gk)
                                     : (hn + (size_t)tIdx[r] * 128 + (gk - 128));
      float4 w = *(const float4*)base;
      if (rowBase + r >= Ei) { w.x = 0.f; w.y = 0.f; w.z = 0.f; w.w = 0.f; }
      H4 p;
      p.h[0] = (_Float16)w.x; p.h[1] = (_Float16)w.y;
      p.h[2] = (_Float16)w.z; p.h[3] = (_Float16)w.w;
      *(uint2*)(As + r * LDSD + q) = p.u;
    }
    stage_Bn(W, 128, 259, colBase, k0, tid, Bs);
    __syncthreads();
    wmma_tile(As, Bs, wave, lane, acc);
    __syncthreads();
  }

  // tail tile: k = 256..287, only 3 valid columns (edge_attr)
  {
    const uint2 z = {0u, 0u};
#pragma unroll
    for (int it = 0; it < 4; ++it) {
      int idx = tid + it * 128;
      int r = idx >> 3, q = (idx & 7) * 4;
      *(uint2*)(As + r * LDSD + q) = z;
    }
    __syncthreads();
    if (tid < TILE_M) {
      int gr = rowBase + tid;
      if (gr < Ei) {
        size_t eb = (size_t)(eoff + gr) * 3;
        As[tid * LDSD + 0] = (_Float16)eattr[eb + 0];
        As[tid * LDSD + 1] = (_Float16)eattr[eb + 1];
        As[tid * LDSD + 2] = (_Float16)eattr[eb + 2];
      }
    }
    stage_Bn(W, 128, 259, colBase, 256, tid, Bs);
    __syncthreads();
    wmma_tile(As, Bs, wave, lane, acc);
  }

  const int colL = lane & 15;
  const int rsel = (lane >> 4) * 8;
#pragma unroll
  for (int ns = 0; ns < 4; ++ns) {
    int gcol = colBase + ns * 16 + colL;
    float bv = bias[gcol];
#pragma unroll
    for (int r = 0; r < 8; ++r) {
      int grow = rowBase + wave * 16 + rsel + r;
      if (grow < Ei) {
        float v = acc[ns][r] + bv;
        Cout[(size_t)grow * 128 + gcol] = v > 0.f ? v : 0.f;
      }
    }
  }
}

// Edge message layer 2 + scatter-add: m = Min(Ei,128) @ W(128,128) + bias,
// then ui[tgt[e], slice + col] += m  (fp32 global atomics)
__global__ void __launch_bounds__(128) edge_msg2_scatter_kernel(
    const float* __restrict__ Min,
    const int* __restrict__ tgt, int eoff, int Ei,
    const float* __restrict__ W, const float* __restrict__ bias,
    float* __restrict__ ui, int slice)
{
  __shared__ __align__(16) _Float16 As[TILE_M * LDSD];
  __shared__ __align__(16) _Float16 Bs[TILE_N * LDSD];
  const int tid = threadIdx.x;
  const int wave = tid >> 5, lane = tid & 31;
  const int rowBase = blockIdx.y * TILE_M;
  const int colBase = blockIdx.x * TILE_N;
  v8f acc[4] = {};
  for (int k0 = 0; k0 < 128; k0 += TILE_K) {
    stage_A(Min, 128, Ei, 128, rowBase, k0, tid, As);
    stage_Bn(W, 128, 128, colBase, k0, tid, Bs);
    __syncthreads();
    wmma_tile(As, Bs, wave, lane, acc);
    __syncthreads();
  }
  const int colL = lane & 15;
  const int rsel = (lane >> 4) * 8;
#pragma unroll
  for (int ns = 0; ns < 4; ++ns) {
    int gcol = colBase + ns * 16 + colL;
    float bv = bias[gcol];
#pragma unroll
    for (int r = 0; r < 8; ++r) {
      int grow = rowBase + wave * 16 + rsel + r;
      if (grow < Ei) {
        int node = tgt[eoff + grow];
        atomicAdd(&ui[(size_t)node * 512 + slice + gcol], acc[ns][r] + bv);
      }
    }
  }
}

// ui[:, 0:128] = hn ; ui[:, 128:512] = 0
__global__ void ui_init_kernel(float* __restrict__ ui, const float* __restrict__ hn, int n) {
  int idx = blockIdx.x * blockDim.x + threadIdx.x;
  if (idx >= n * 512) return;
  int row = idx >> 9, c = idx & 511;
  ui[idx] = (c < 128) ? hn[row * 128 + c] : 0.f;
}

__global__ void fill_zero_kernel(float* __restrict__ p, int n) {
  int i = blockIdx.x * blockDim.x + threadIdx.x;
  if (i < n) p[i] = 0.f;
}

// out = gate*upd + (1-gate)*hn, then LayerNorm over 128 cols -> hn (in place)
__global__ void __launch_bounds__(128) combine_ln_kernel(
    const float* __restrict__ gate, const float* __restrict__ upd,
    float* __restrict__ hn, const float* __restrict__ g, const float* __restrict__ b)
{
  const int row = blockIdx.x, c = threadIdx.x;
  __shared__ float red[128];
  size_t base = (size_t)row * 128 + c;
  float gt = gate[base], up = upd[base], hv = hn[base];
  float o = gt * up + (1.f - gt) * hv;
  red[c] = o; __syncthreads();
  for (int s = 64; s > 0; s >>= 1) { if (c < s) red[c] += red[c + s]; __syncthreads(); }
  float m = red[0] / 128.f; __syncthreads();
  float d = o - m;
  red[c] = d * d; __syncthreads();
  for (int s = 64; s > 0; s >>= 1) { if (c < s) red[c] += red[c + s]; __syncthreads(); }
  float var = red[0] / 128.f;
  hn[base] = d * rsqrtf(var + 1e-5f) * g[c] + b[c];
}

// LSTM cell: gates(n,4*hd) [i|f|g|o]; c = sig(f)*c + sig(i)*tanh(g); h = sig(o)*tanh(c)
__global__ void lstm_cell_kernel(const float* __restrict__ gates,
                                 float* __restrict__ h, float* __restrict__ c,
                                 float* __restrict__ outs, int n, int hd)
{
  int idx = blockIdx.x * blockDim.x + threadIdx.x;
  if (idx >= n * hd) return;
  int row = idx / hd, k = idx - row * hd;
  const float* gr = gates + (size_t)row * 4 * hd;
  float gi = gr[k], gf = gr[hd + k], gg = gr[2 * hd + k], go = gr[3 * hd + k];
  float si = 1.f / (1.f + __expf(-gi));
  float sf = 1.f / (1.f + __expf(-gf));
  float so = 1.f / (1.f + __expf(-go));
  float cc = sf * c[idx] + si * tanhf(gg);
  float hh = so * tanhf(cc);
  c[idx] = cc; h[idx] = hh;
  if (outs) outs[idx] = hh;
}

// ---------------------------------------------------------------------------
extern "C" void kernel_launch(void* const* d_in, const int* in_sizes, int n_in,
                              void* d_out, int out_size, void* d_ws, size_t ws_size,
                              hipStream_t stream) {
  (void)in_sizes; (void)n_in; (void)out_size; (void)ws_size;

  constexpr int NN = 25000, Fdim = 16, Hd = 128, THd = 64, Ld = 4, Td = 4;
  constexpr int EL = 150000, EM = 75000, ELG = 37500, Etot = EL + EM + ELG;
  const int offs[4] = {0, EL, EL + EM, Etot};
  const int cnts[3] = {EL, EM, ELG};

  const float* x      = (const float*)d_in[0];
  const int*   eidx   = (const int*)d_in[1];
  const float* eattr  = (const float*)d_in[2];
  const float* enc_w1 = (const float*)d_in[6];
  const float* enc_b1 = (const float*)d_in[7];
  const float* enc_w2 = (const float*)d_in[8];
  const float* enc_b2 = (const float*)d_in[9];
  const float* msg_w1 = (const float*)d_in[10];
  const float* msg_b1 = (const float*)d_in[11];
  const float* msg_w2 = (const float*)d_in[12];
  const float* msg_b2 = (const float*)d_in[13];
  const float* gate_w = (const float*)d_in[14];
  const float* gate_b = (const float*)d_in[15];
  const float* upd_w1 = (const float*)d_in[16];
  const float* upd_b1 = (const float*)d_in[17];
  const float* upd_w2 = (const float*)d_in[18];
  const float* upd_b2 = (const float*)d_in[19];
  const float* ln_g   = (const float*)d_in[20];
  const float* ln_b   = (const float*)d_in[21];
  const float* wih0   = (const float*)d_in[22];
  const float* whh0   = (const float*)d_in[23];
  const float* bih0   = (const float*)d_in[24];
  const float* bhh0   = (const float*)d_in[25];
  const float* wih1   = (const float*)d_in[26];
  const float* whh1   = (const float*)d_in[27];
  const float* bih1   = (const float*)d_in[28];
  const float* bhh1   = (const float*)d_in[29];
  const float* out_w1 = (const float*)d_in[30];
  const float* out_b1 = (const float*)d_in[31];
  const float* out_w2 = (const float*)d_in[32];
  const float* out_b2 = (const float*)d_in[33];

  const int* src = eidx;
  const int* tgt = eidx + Etot;

  // workspace carve-out
  float* wsf = (float*)d_ws;
  size_t off = 0;
  auto alloc = [&](size_t n) { float* p = wsf + off; off += n; return p; };
  float* hn_all  = alloc((size_t)Td * NN * Hd);   // node states, all 4 graphs
  float* mtmp    = alloc((size_t)EL * Hd);        // edge msg intermediate (also encoder h1)
  float* ui      = alloc((size_t)NN * 4 * Hd);    // [hn|agg0|agg1|agg2]
  float* gatebuf = alloc((size_t)NN * Hd);
  float* u1      = alloc((size_t)NN * Hd);
  float* updbuf  = alloc((size_t)NN * Hd);
  float* gates   = alloc((size_t)NN * 4 * THd);
  float* o1seq   = alloc((size_t)Td * NN * THd);
  float* h0 = alloc((size_t)NN * THd);
  float* c0 = alloc((size_t)NN * THd);
  float* h1 = alloc((size_t)NN * THd);
  float* c1 = alloc((size_t)NN * THd);
  float* y1 = alloc((size_t)NN * THd);

  auto gemm = [&](const float* A, int lda, const float* B, const float* bias,
                  float* C, int ldc, int M, int N, int K, int transB, int act, int beta) {
    dim3 grid((N + 63) / 64, (M + 63) / 64);
    if (transB)
      gemm_kernel<1><<<grid, dim3(128), 0, stream>>>(A, lda, B, bias, C, ldc, M, N, K, act, beta);
    else
      gemm_kernel<0><<<grid, dim3(128), 0, stream>>>(A, lda, B, bias, C, ldc, M, N, K, act, beta);
  };

  // ---- encoder: (100000,16) -> relu -> (100000,64) -> relu -> (100000,128)
  const int NT = Td * NN;
  gemm(x, Fdim, enc_w1, enc_b1, mtmp, 64, NT, 64, Fdim, 0, 1, 0);
  gemm(mtmp, 64, enc_w2, enc_b2, hn_all, Hd, NT, Hd, 64, 0, 1, 0);

  // ---- spatial message passing: per graph, per layer
  for (int g = 0; g < Td; ++g) {
    float* hn = hn_all + (size_t)g * NN * Hd;
    for (int l = 0; l < Ld; ++l) {
      ui_init_kernel<<<(NN * 512 + 255) / 256, 256, 0, stream>>>(ui, hn, NN);
      for (int i = 0; i < 3; ++i) {
        int eoff = offs[i], Ei = cnts[i];
        const float* W1 = msg_w1 + (size_t)(l * 3 + i) * 259 * Hd;
        const float* B1 = msg_b1 + (size_t)(l * 3 + i) * Hd;
        const float* W2 = msg_w2 + (size_t)(l * 3 + i) * Hd * Hd;
        const float* B2 = msg_b2 + (size_t)(l * 3 + i) * Hd;
        dim3 eg(2, (Ei + 63) / 64);
        edge_msg1_kernel<<<eg, dim3(128), 0, stream>>>(hn, src, tgt, eattr, eoff, Ei, W1, B1, mtmp);
        edge_msg2_scatter_kernel<<<eg, dim3(128), 0, stream>>>(mtmp, tgt, eoff, Ei, W2, B2, ui, Hd * (i + 1));
      }
      gemm(ui, 4 * Hd, gate_w + (size_t)l * 4 * Hd * Hd, gate_b + l * Hd, gatebuf, Hd, NN, Hd, 4 * Hd, 0, 2, 0);
      gemm(ui, 4 * Hd, upd_w1 + (size_t)l * 4 * Hd * Hd, upd_b1 + l * Hd, u1, Hd, NN, Hd, 4 * Hd, 0, 1, 0);
      gemm(u1, Hd, upd_w2 + (size_t)l * Hd * Hd, upd_b2 + l * Hd, updbuf, Hd, NN, Hd, Hd, 0, 0, 0);
      combine_ln_kernel<<<NN, 128, 0, stream>>>(gatebuf, updbuf, hn, ln_g + l * Hd, ln_b + l * Hd);
    }
  }

  // ---- LSTM layer 0 (seq[t] = hn_all graph t, since B=1)
  fill_zero_kernel<<<(NN * THd + 255) / 256, 256, 0, stream>>>(h0, NN * THd);
  fill_zero_kernel<<<(NN * THd + 255) / 256, 256, 0, stream>>>(c0, NN * THd);
  for (int t = 0; t < Td; ++t) {
    const float* xt = hn_all + (size_t)t * NN * Hd;
    gemm(xt, Hd, wih0, bih0, gates, 4 * THd, NN, 4 * THd, Hd, 1, 0, 0);
    gemm(h0, THd, whh0, bhh0, gates, 4 * THd, NN, 4 * THd, THd, 1, 0, 1);
    lstm_cell_kernel<<<(NN * THd + 255) / 256, 256, 0, stream>>>(gates, h0, c0, o1seq + (size_t)t * NN * THd, NN, THd);
  }
  // ---- LSTM layer 1
  fill_zero_kernel<<<(NN * THd + 255) / 256, 256, 0, stream>>>(h1, NN * THd);
  fill_zero_kernel<<<(NN * THd + 255) / 256, 256, 0, stream>>>(c1, NN * THd);
  for (int t = 0; t < Td; ++t) {
    gemm(o1seq + (size_t)t * NN * THd, THd, wih1, bih1, gates, 4 * THd, NN, 4 * THd, THd, 1, 0, 0);
    gemm(h1, THd, whh1, bhh1, gates, 4 * THd, NN, 4 * THd, THd, 1, 0, 1);
    lstm_cell_kernel<<<(NN * THd + 255) / 256, 256, 0, stream>>>(gates, h1, c1, nullptr, NN, THd);
  }

  // ---- output head: relu((25000,64)@(64,64)) @ (64,4)
  gemm(h1, THd, out_w1, out_b1, y1, THd, NN, THd, THd, 0, 1, 0);
  gemm(y1, THd, out_w2, out_b2, (float*)d_out, 4, NN, 4, THd, 0, 0, 0);
}
